// NonLocalBlock_274877907241
// MI455X (gfx1250) — matrix operand order repaired and checked by reading, not compiled, not run on hardware
//
#include <hip/hip_runtime.h>
#include <hip/hip_bf16.h>

typedef __attribute__((ext_vector_type(16))) _Float16 v16h;
typedef __attribute__((ext_vector_type(8)))  _Float16 v8h;
typedef __attribute__((ext_vector_type(4)))  _Float16 v4h;
typedef __attribute__((ext_vector_type(8)))  float    v8f;
typedef __attribute__((ext_vector_type(4)))  float    v4f;
typedef __attribute__((ext_vector_type(4)))  int      v4i;

#define DEVINL __device__ __forceinline__

static constexpr int BATCH = 4;
static constexpr int C     = 256;
static constexpr int NSP   = 4096;   // 64*64
static constexpr int G     = 32;
static constexpr int CG    = C / G;  // 8

#if defined(__AMDGCN__) && __has_builtin(__builtin_amdgcn_global_load_async_to_lds_b128)
#define HAVE_ASYNC_LDS 1
#endif

#ifdef HAVE_ASYNC_LDS
typedef __attribute__((address_space(1))) v4i* gv4i_p;
typedef __attribute__((address_space(3))) v4i* lv4i_p;
#endif

// straight 16-byte global -> LDS copy; async (ASYNCcnt) when available
DEVINL void copy16_to_lds(const _Float16* g, _Float16* l) {
#ifdef HAVE_ASYNC_LDS
  __builtin_amdgcn_global_load_async_to_lds_b128((gv4i_p)g, (lv4i_p)l, 0, 0);
#else
  *(v8h*)l = *(const v8h*)g;
#endif
}

DEVINL void async_copy_wait() {
#ifdef HAVE_ASYNC_LDS
  asm volatile("s_wait_asynccnt 0x0" ::: "memory");
#endif
}

DEVINL v8f zero8() {
  v8f z;
  #pragma unroll
  for (int i = 0; i < 8; ++i) z[i] = 0.0f;
  return z;
}

DEVINL v8f wmma_f16(v16h a, v16h b, v8f c) {
  return __builtin_amdgcn_wmma_f32_16x16x32_f16(false, a, false, b, (short)0, c, false, false);
}

// A fragment (16x32, M x K), row-major LDS tile [m][k] with pitch `pitch` halfs.
// lanes 0-15: M=lane, halfs0-7=K0-7, halfs8-15=K16-23
// lanes16-31: M=lane-16, halfs0-7=K8-15, halfs8-15=K24-31
DEVINL v16h ld_afrag(const _Float16* base, int pitch, int lane) {
  int m  = lane & 15;
  int ko = (lane >> 4) << 3;  // 0 or 8
  v8h lo = *(const v8h*)(base + m * pitch + ko);
  v8h hi = *(const v8h*)(base + m * pitch + 16 + ko);
  v16h r;
  #pragma unroll
  for (int i = 0; i < 8; ++i) { r[i] = lo[i]; r[i + 8] = hi[i]; }
  return r;
}

// B fragment (32x16, K x N), from K-transposed LDS tile [n][k] with pitch halfs.
// lanes 0-15: N=lane, halfs0-7=K0-7, halfs8-15=K16-23 ; lanes16-31: N=lane-16, halfs0-7=K8-15, halfs8-15=K24-31
DEVINL v16h ld_bfrag(const _Float16* baseT, int pitch, int lane) {
  int n  = lane & 15;
  int kb = (lane >> 4) << 3;  // 0 or 8
  const _Float16* p = baseT + n * pitch + kb;
  v8h lo = *(const v8h*)p;
  v8h hi = *(const v8h*)(p + 16);
  v16h r;
  #pragma unroll
  for (int i = 0; i < 8; ++i) { r[i] = lo[i]; r[i + 8] = hi[i]; }
  return r;
}

// ---------------- GroupNorm pass 1: per (b,g) mean / rstd ----------------
__global__ __launch_bounds__(256) void gn_stats_kernel(const float* __restrict__ x,
                                                       float* __restrict__ stats) {
  int bg = blockIdx.x;                           // 0..127  (b*G+g)
  const v4f* p = (const v4f*)(x + (size_t)bg * (CG * NSP));
  float s = 0.f, s2 = 0.f;
  for (int i = threadIdx.x; i < CG * NSP / 4; i += 256) {
    v4f v = p[i];
    s  += v.x + v.y + v.z + v.w;
    s2 += v.x * v.x + v.y * v.y + v.z * v.z + v.w * v.w;
  }
  __shared__ float ls[256], lq[256];
  ls[threadIdx.x] = s; lq[threadIdx.x] = s2;
  __syncthreads();
  for (int off = 128; off > 0; off >>= 1) {
    if (threadIdx.x < off) { ls[threadIdx.x] += ls[threadIdx.x + off];
                             lq[threadIdx.x] += lq[threadIdx.x + off]; }
    __syncthreads();
  }
  if (threadIdx.x == 0) {
    float inv  = 1.0f / (float)(CG * NSP);
    float mean = ls[0] * inv;
    float var  = lq[0] * inv - mean * mean;
    stats[2 * bg]     = mean;
    stats[2 * bg + 1] = rsqrtf(var + 1e-6f);
  }
}

// ---------------- GroupNorm apply + fp32 -> f16 ----------------
__global__ __launch_bounds__(256) void gn_apply_kernel(const float* __restrict__ x,
                                                       const float* __restrict__ gsc,
                                                       const float* __restrict__ gbi,
                                                       const float* __restrict__ stats,
                                                       _Float16* __restrict__ hn) {
  int bc = blockIdx.x;            // b*C + c
  int c  = bc & (C - 1);
  int b  = bc >> 8;
  int g  = c >> 3;
  float mean = stats[2 * (b * G + g)];
  float rstd = stats[2 * (b * G + g) + 1];
  float sc   = gsc[c] * rstd;
  float bi   = gbi[c] - mean * sc;
  const v4f* px = (const v4f*)(x + (size_t)bc * NSP);
  v4h* ph = (v4h*)(hn + (size_t)bc * NSP);
  for (int i = threadIdx.x; i < NSP / 4; i += 256) {
    v4f v = px[i];
    v4h o;
    o[0] = (_Float16)(v.x * sc + bi);
    o[1] = (_Float16)(v.y * sc + bi);
    o[2] = (_Float16)(v.z * sc + bi);
    o[3] = (_Float16)(v.w * sc + bi);
    ph[i] = o;
  }
}

// ---------------- weights fp32 -> f16 (wq,wk,wv,wproj contiguous) ----------------
__global__ __launch_bounds__(256) void cvt_w_kernel(const float* __restrict__ wq,
                                                    const float* __restrict__ wk,
                                                    const float* __restrict__ wv,
                                                    const float* __restrict__ wp,
                                                    _Float16* __restrict__ o) {
  int idx = blockIdx.x * 256 + threadIdx.x;      // 0..262143
  int m = idx >> 16, r = idx & 65535;
  const float* s = (m == 0) ? wq : (m == 1) ? wk : (m == 2) ? wv : wp;
  o[idx] = (_Float16)s[r];
}

// ---------------- fused q/k/v GEMM (WMMA) ----------------
// q[b][d][n] = (sum_c wq[d][c]*hn[b][c][n] + bq[d]) * 1/16   (attn scale folded)
__global__ __launch_bounds__(128) void qkv_kernel(const _Float16* __restrict__ hn,
                                                  const _Float16* __restrict__ wqh,
                                                  const _Float16* __restrict__ wkh,
                                                  const _Float16* __restrict__ wvh,
                                                  const float* __restrict__ bq,
                                                  const float* __restrict__ bk,
                                                  const float* __restrict__ bv,
                                                  _Float16* __restrict__ q,
                                                  _Float16* __restrict__ k,
                                                  _Float16* __restrict__ v) {
  __shared__ _Float16 hsT[64][32];      // [n][c-chunk]  (B operand, K-transposed)
  __shared__ _Float16 ws[3][64][32];    // [mat][d][c-chunk]  (A operand)
  int n0 = blockIdx.x * 64;
  int d0 = blockIdx.y * 64;
  int b  = blockIdx.z;
  int lane = threadIdx.x & 31, w = threadIdx.x >> 5;
  int d0w = d0 + w * 16;

  v8f acc[3][4];
  #pragma unroll
  for (int m = 0; m < 3; ++m)
    #pragma unroll
    for (int t = 0; t < 4; ++t) acc[m][t] = zero8();

  for (int ks = 0; ks < 8; ++ks) {
    int c0 = ks * 32;
    __syncthreads();
    // stage 3 weight tiles row-major (straight copy -> async to LDS)
    #pragma unroll
    for (int pass = 0; pass < 6; ++pass) {
      int chunk = threadIdx.x + pass * 128;          // 0..767
      int mat = chunk >> 8;
      int rr  = chunk & 255;
      int dd  = rr >> 2, cb = (rr & 3) * 8;
      const _Float16* wsrc = (mat == 0) ? wqh : (mat == 1) ? wkh : wvh;
      copy16_to_lds(wsrc + (size_t)(d0 + dd) * C + c0 + cb, &ws[mat][dd][cb]);
    }
    // stage hn tile transposed: hsT[n][c] = hn[b][c0+c][n0+n]  (2B scatter, manual)
    #pragma unroll
    for (int pass = 0; pass < 2; ++pass) {
      int chunk = threadIdx.x + pass * 128;          // 0..255
      int cc = chunk >> 3, nb = (chunk & 7) * 8;
      v8h s = *(const v8h*)(hn + ((size_t)(b * C + c0 + cc)) * NSP + n0 + nb);
      #pragma unroll
      for (int j = 0; j < 8; ++j) hsT[nb + j][cc] = s[j];
    }
    async_copy_wait();
    __syncthreads();

    v16h bf[4];
    #pragma unroll
    for (int t = 0; t < 4; ++t) bf[t] = ld_bfrag(&hsT[t * 16][0], 32, lane);
    #pragma unroll
    for (int mat = 0; mat < 3; ++mat) {
      v16h a = ld_afrag(&ws[mat][w * 16][0], 32, lane);
      #pragma unroll
      for (int t = 0; t < 4; ++t) acc[mat][t] = wmma_f16(a, bf[t], acc[mat][t]);
    }
  }

  // store: D tile layout — d = d0w + r + 8*(lane>>4), n = n0 + t*16 + (lane&15)
  int drow = d0w + 8 * (lane >> 4);
  #pragma unroll
  for (int mat = 0; mat < 3; ++mat) {
    _Float16* op = (mat == 0) ? q : (mat == 1) ? k : v;
    const float* bp = (mat == 0) ? bq : (mat == 1) ? bk : bv;
    float scl = (mat == 0) ? 0.0625f : 1.0f;
    float bias8[8];
    #pragma unroll
    for (int r = 0; r < 8; ++r) bias8[r] = bp[drow + r];
    #pragma unroll
    for (int t = 0; t < 4; ++t) {
      int n = n0 + t * 16 + (lane & 15);
      #pragma unroll
      for (int r = 0; r < 8; ++r) {
        float vv = (acc[mat][t][r] + bias8[r]) * scl;
        op[((size_t)(b * C + drow + r)) * NSP + n] = (_Float16)vv;
      }
    }
  }
}

// ---------------- flash attention (WMMA, online softmax) ----------------
__global__ __launch_bounds__(128) void attn_kernel(const _Float16* __restrict__ q,
                                                   const _Float16* __restrict__ kmat,
                                                   const _Float16* __restrict__ vmat,
                                                   _Float16* __restrict__ aout) {
  __shared__ _Float16 smem[64 * 256];   // 32KB: first qT[64][256], then kT/vt
  __shared__ _Float16 ps[4][16][32];    // per-wave P bounce (C-layout -> A-layout)
  int i0 = blockIdx.x * 64;
  int b  = blockIdx.y;
  int lane = threadIdx.x & 31, w = threadIdx.x >> 5;
  int i0w = i0 + w * 16;

  _Float16(*qT)[256] = (_Float16(*)[256])smem;
  // stage qT[i][c] = q[b][c][i0+i]
  #pragma unroll
  for (int pass = 0; pass < 16; ++pass) {
    int chunk = threadIdx.x + pass * 128;            // 0..2047
    int cc = chunk >> 3, ib = (chunk & 7) * 8;
    v8h s = *(const v8h*)(q + ((size_t)(b * C + cc)) * NSP + i0 + ib);
    #pragma unroll
    for (int j = 0; j < 8; ++j) qT[ib + j][cc] = s[j];
  }
  __syncthreads();
  // preload Q A-fragments for all 8 K-steps (kept in VGPRs for the whole j loop)
  v16h qa[8];
  #pragma unroll
  for (int ks = 0; ks < 8; ++ks) qa[ks] = ld_afrag(&qT[w * 16][ks * 32], 256, lane);
  __syncthreads();   // qT space will be recycled for kT/vt

  _Float16(*kT)[256] = (_Float16(*)[256])smem;            // [32 j][256 c]
  _Float16(*vt)[32]  = (_Float16(*)[32])(smem + 32 * 256); // [256 c][32 j]

  v8f Oacc[16];
  #pragma unroll
  for (int t = 0; t < 16; ++t) Oacc[t] = zero8();
  float mrow[8], lrow[8];
  #pragma unroll
  for (int r = 0; r < 8; ++r) { mrow[r] = -1e30f; lrow[r] = 0.0f; }

  for (int j0 = 0; j0 < NSP; j0 += 32) {
    __syncthreads();
    // stage V straight: vt[c][j] = v[b][c][j0+j]   (async -> LDS)
    #pragma unroll
    for (int pass = 0; pass < 8; ++pass) {
      int chunk = threadIdx.x + pass * 128;
      int cc = chunk >> 2, jb = (chunk & 3) * 8;
      copy16_to_lds(vmat + ((size_t)(b * C + cc)) * NSP + j0 + jb, &vt[cc][jb]);
    }
    // stage K transposed: kT[j][c] = k[b][c][j0+j]  (2B scatter, manual)
    #pragma unroll
    for (int pass = 0; pass < 8; ++pass) {
      int chunk = threadIdx.x + pass * 128;          // 0..1023
      int cc = chunk >> 2, jb = (chunk & 3) * 8;
      v8h s = *(const v8h*)(kmat + ((size_t)(b * C + cc)) * NSP + j0 + jb);
      #pragma unroll
      for (int j = 0; j < 8; ++j) kT[jb + j][cc] = s[j];
    }
    // prefetch next step's K/V rows into cache while this step computes
    if (j0 + 32 < NSP) {
      int cc = threadIdx.x & 255;
      __builtin_prefetch(kmat + ((size_t)(b * C + cc)) * NSP + j0 + 32, 0, 1);
      __builtin_prefetch(vmat + ((size_t)(b * C + cc)) * NSP + j0 + 32, 0, 1);
    }
    async_copy_wait();
    __syncthreads();

    // S tiles: S[i][j] = sum_c q[c][i]*k[c][j]  (scale already folded into q)
    v8f s0 = zero8(), s1 = zero8();
    #pragma unroll
    for (int ks = 0; ks < 8; ++ks) {
      v16h b0 = ld_bfrag(&kT[0][ks * 32], 256, lane);
      v16h b1 = ld_bfrag(&kT[16][ks * 32], 256, lane);
      s0 = wmma_f16(qa[ks], b0, s0);
      s1 = wmma_f16(qa[ks], b1, s1);
    }

    // online softmax; row i = r + 8*(lane>>4), j = lane&15 (+16 for s1)
    float alpha[8];
    #pragma unroll
    for (int r = 0; r < 8; ++r) {
      float mx = fmaxf(s0[r], s1[r]);
      #pragma unroll
      for (int m = 8; m > 0; m >>= 1) mx = fmaxf(mx, __shfl_xor(mx, m, 32));
      float mn = fmaxf(mrow[r], mx);
      alpha[r] = __expf(mrow[r] - mn);
      mrow[r]  = mn;
      float p0 = __expf(s0[r] - mn);
      float p1 = __expf(s1[r] - mn);
      s0[r] = p0; s1[r] = p1;
      float rs = p0 + p1;
      #pragma unroll
      for (int m = 8; m > 0; m >>= 1) rs += __shfl_xor(rs, m, 32);
      lrow[r] = lrow[r] * alpha[r] + rs;
    }
    #pragma unroll
    for (int t = 0; t < 16; ++t)
      #pragma unroll
      for (int r = 0; r < 8; ++r) Oacc[t][r] *= alpha[r];

    // bounce P through LDS to get A-layout fragment
    #pragma unroll
    for (int r = 0; r < 8; ++r) {
      int irow = r + 8 * (lane >> 4);
      ps[w][irow][lane & 15]        = (_Float16)s0[r];
      ps[w][irow][16 + (lane & 15)] = (_Float16)s1[r];
    }
    asm volatile("s_wait_dscnt 0x0" ::: "memory");
    v16h pa = ld_afrag(&ps[w][0][0], 32, lane);

    // O[i][c] += sum_j P[i][j] * v[c][j]
    #pragma unroll
    for (int t = 0; t < 16; ++t) {
      v16h bf = ld_bfrag(&vt[t * 16][0], 32, lane);
      Oacc[t] = wmma_f16(pa, bf, Oacc[t]);
    }
  }

  // epilogue: divide by row sums; O tile: i = i0w + r + 8*(lane>>4), c = t*16 + (lane&15)
  float rinv[8];
  #pragma unroll
  for (int r = 0; r < 8; ++r) rinv[r] = 1.0f / lrow[r];
  int ibase = i0w + 8 * (lane >> 4);
  #pragma unroll
  for (int t = 0; t < 16; ++t) {
    int c = t * 16 + (lane & 15);
    v8h o;
    #pragma unroll
    for (int r = 0; r < 8; ++r) o[r] = (_Float16)(Oacc[t][r] * rinv[r]);
    *(v8h*)(aout + ((size_t)(b * C + c)) * NSP + ibase) = o;
  }
}

// ---------------- proj GEMM + bias + residual (fp32 out) ----------------
__global__ __launch_bounds__(128) void proj_kernel(const _Float16* __restrict__ ain,
                                                   const _Float16* __restrict__ wph,
                                                   const float* __restrict__ bp,
                                                   const float* __restrict__ x,
                                                   float* __restrict__ out) {
  __shared__ _Float16 asT[64][32];
  __shared__ _Float16 wps[64][32];
  int n0 = blockIdx.x * 64;
  int d0 = blockIdx.y * 64;
  int b  = blockIdx.z;
  int lane = threadIdx.x & 31, w = threadIdx.x >> 5;
  int d0w = d0 + w * 16;

  v8f acc[4];
  #pragma unroll
  for (int t = 0; t < 4; ++t) acc[t] = zero8();

  for (int ks = 0; ks < 8; ++ks) {
    int c0 = ks * 32;
    __syncthreads();
    // weight tile: straight copy -> async to LDS
    #pragma unroll
    for (int pass = 0; pass < 2; ++pass) {
      int chunk = threadIdx.x + pass * 128;          // 0..255
      int dd = chunk >> 2, cb = (chunk & 3) * 8;
      copy16_to_lds(wph + (size_t)(d0 + dd) * C + c0 + cb, &wps[dd][cb]);
    }
    // activation tile transposed (2B scatter, manual)
    #pragma unroll
    for (int pass = 0; pass < 2; ++pass) {
      int chunk = threadIdx.x + pass * 128;
      int cc = chunk >> 3, nb = (chunk & 7) * 8;
      v8h s = *(const v8h*)(ain + ((size_t)(b * C + c0 + cc)) * NSP + n0 + nb);
      #pragma unroll
      for (int j = 0; j < 8; ++j) asT[nb + j][cc] = s[j];
    }
    async_copy_wait();
    __syncthreads();
    v16h a = ld_afrag(&wps[w * 16][0], 32, lane);
    #pragma unroll
    for (int t = 0; t < 4; ++t) {
      v16h bf = ld_bfrag(&asT[t * 16][0], 32, lane);
      acc[t] = wmma_f16(a, bf, acc[t]);
    }
  }

  int drow = d0w + 8 * (lane >> 4);
  float bias8[8];
  #pragma unroll
  for (int r = 0; r < 8; ++r) bias8[r] = bp[drow + r];
  #pragma unroll
  for (int t = 0; t < 4; ++t) {
    int n = n0 + t * 16 + (lane & 15);
    #pragma unroll
    for (int r = 0; r < 8; ++r) {
      size_t idx = ((size_t)(b * C + drow + r)) * NSP + n;
      out[idx] = acc[t][r] + bias8[r] + x[idx];
    }
  }
}

extern "C" void kernel_launch(void* const* d_in, const int* in_sizes, int n_in,
                              void* d_out, int out_size, void* d_ws, size_t ws_size,
                              hipStream_t stream) {
  const float* x    = (const float*)d_in[0];
  const float* gsc  = (const float*)d_in[1];
  const float* gbi  = (const float*)d_in[2];
  const float* wq   = (const float*)d_in[3];
  const float* bq   = (const float*)d_in[4];
  const float* wk   = (const float*)d_in[5];
  const float* bk   = (const float*)d_in[6];
  const float* wv   = (const float*)d_in[7];
  const float* bv   = (const float*)d_in[8];
  const float* wp   = (const float*)d_in[9];
  const float* bp   = (const float*)d_in[10];
  float* out = (float*)d_out;

  char* w = (char*)d_ws;
  float*    stats = (float*)w;                              // 1 KB
  _Float16* wqh   = (_Float16*)(w + 4096);
  _Float16* wkh   = wqh + 65536;
  _Float16* wvh   = wkh + 65536;
  _Float16* wph   = wvh + 65536;
  _Float16* hn    = wph + 65536;                            // 8 MB
  size_t    full  = (size_t)BATCH * C * NSP;
  _Float16* qb    = hn + full;
  _Float16* kb    = qb + full;
  _Float16* vb    = kb + full;
  _Float16* ab    = vb + full;

  gn_stats_kernel<<<BATCH * G, 256, 0, stream>>>(x, stats);
  gn_apply_kernel<<<BATCH * C, 256, 0, stream>>>(x, gsc, gbi, stats, hn);
  cvt_w_kernel<<<(4 * C * C) / 256, 256, 0, stream>>>(wq, wk, wv, wp, wqh);
  qkv_kernel<<<dim3(NSP / 64, C / 64, BATCH), 128, 0, stream>>>(
      hn, wqh, wkh, wvh, bq, bk, bv, qb, kb, vb);
  attn_kernel<<<dim3(NSP / 64, BATCH), 128, 0, stream>>>(qb, kb, vb, ab);
  proj_kernel<<<dim3(NSP / 64, C / 64, BATCH), 128, 0, stream>>>(ab, wph, bp, x, out);
}
